// EfficientSelfAttentionWithRelPos_27633819582732
// MI455X (gfx1250) — compile-verified, matrix operand order checked by reading, not run on hardware
//
#include <hip/hip_runtime.h>
#include <cmath>

// MI455X (gfx1250) wave32 WMMA implementation of windowed attention with
// relative position bias. d_out = [out (B*N*128 f32)] ++ [attn (B*4*N*N f32)].
//
// Pipeline:
//   k0: f32 -> f16 conversion of x, q_w, kv_w, proj_w
//   k1: fused QKV projection GEMM (v_wmma_f32_16x16x32_f16), writes
//       q16 [b,h,n,32] (scaled by 1/sqrt(32)), k16 [b,h,n,32], vT16 [b,h,32,n]
//   k2: per 16-row query tile: two-pass softmax
//       pass1: S = Q*K^T + bias -> lane-local online (m,l), one cross-lane
//              merge at the end (no per-element shuffles)
//       pass2: recompute S, p = exp(s-m)/l, write attn f32 once, O += P*V
//              (P relaid out C->A via wave-private LDS + s_wait_dscnt)
//   k3: output projection GEMM -> out f32
//
// Bias gather index is fully strength-reduced: idx = c_r - d_k with
// c_r = (qh+H-1)*(2W-1) + qw+W-1 per query row and d_k = kh*(2W-1)+kw
// maintained incrementally (no integer division in the K loops).

typedef __attribute__((ext_vector_type(16))) _Float16 v16h;
typedef __attribute__((ext_vector_type(8)))  _Float16 v8h;
typedef __attribute__((ext_vector_type(8)))  float    v8f;

__device__ __forceinline__ v8f wmma16(v16h a, v16h b, v8f c) {
  // (neg_a, A, neg_b, B, c_mod, C, reuse_a, reuse_b)
  return __builtin_amdgcn_wmma_f32_16x16x32_f16(false, a, false, b, (short)0, c, false, false);
}

// Row-major f16 fragment loader; identical per-lane layout for A and for B
// when computing D = A * B^T (ISA 7.12.2).
__device__ __forceinline__ v16h load_frag(const _Float16* __restrict__ base,
                                          int ld, int row0, int col0, int lane) {
  int m = lane & 15, hi = lane >> 4;
  const _Float16* p = base + (size_t)(row0 + m) * (size_t)ld + col0 + hi * 8;
  v8h lo = *(const v8h*)p;
  v8h hv = *(const v8h*)(p + 16);
  v16h r;
#pragma unroll
  for (int i = 0; i < 8; ++i) { r[i] = lo[i]; r[i + 8] = hv[i]; }
  return r;
}

// ---------------- k0: f32 -> f16 conversions ----------------
__global__ void __launch_bounds__(256)
convert_kernel(const float* __restrict__ x, const float* __restrict__ qw,
               const float* __restrict__ kvw, const float* __restrict__ pw,
               _Float16* xh, _Float16* qwh, _Float16* kvwh, _Float16* pwh,
               int n_x, int n_qw, int n_kvw, int n_pw) {
  int i = blockIdx.x * blockDim.x + threadIdx.x;
  if (i < n_x) { xh[i] = (_Float16)x[i]; return; }
  int j = i - n_x;
  if (j < n_qw) { qwh[j] = (_Float16)qw[j]; return; }
  j -= n_qw;
  if (j < n_kvw) { kvwh[j] = (_Float16)kvw[j]; return; }
  j -= n_kvw;
  if (j < n_pw) { pwh[j] = (_Float16)pw[j]; }
}

// ---------------- k1: fused QKV projection ----------------
__global__ void __launch_bounds__(256)
qkv_kernel(const _Float16* __restrict__ xh, const _Float16* __restrict__ qw,
           const _Float16* __restrict__ kvw, const float* __restrict__ qb,
           const float* __restrict__ kvb, _Float16* __restrict__ q16,
           _Float16* __restrict__ k16, _Float16* __restrict__ vT,
           int BN, int N, int tiles_q, int tiles_total) {
  int lane = threadIdx.x & 31;
  int gw = (blockIdx.x * blockDim.x + threadIdx.x) >> 5;   // wave-uniform tile id
  if (gw >= tiles_total) return;                            // whole wave exits

  int tile_m, tile_n; const _Float16* wmat; bool isQ;
  if (gw < tiles_q) { isQ = true;  tile_m = gw >> 3; tile_n = gw & 7;  wmat = qw; }
  else { int t = gw - tiles_q; isQ = false; tile_m = t >> 4; tile_n = t & 15; wmat = kvw; }
  int row0 = tile_m * 16, col0 = tile_n * 16;

  v8f acc = {0.f, 0.f, 0.f, 0.f, 0.f, 0.f, 0.f, 0.f};
#pragma unroll
  for (int kc = 0; kc < 4; ++kc) {
    v16h a = load_frag(xh,   128, row0, kc * 32, lane);     // A: x rows
    v16h b = load_frag(wmat, 128, col0, kc * 32, lane);     // B: W rows (== W^T cols)
    acc = wmma16(a, b, acc);
  }

  int hi = lane >> 4, nlo = lane & 15;
  int o = col0 + nlo;
  int b_ = row0 / N;                 // 16-row tile never straddles a batch
  int base_n = row0 - b_ * N;

  if (isQ) {
    const float scale = 0.17677669529663689f;  // 1/sqrt(32), folded into Q
    float bias = qb[o];
    int head = o >> 5, d = o & 31;
#pragma unroll
    for (int r = 0; r < 8; ++r) {
      int nn = base_n + r + 8 * hi;
      q16[(((size_t)(b_ * 4 + head)) * N + nn) * 32 + d] = (_Float16)((acc[r] + bias) * scale);
    }
  } else {
    float bias = kvb[o];
    if (o < 128) {                                   // K head-major rows
      int head = o >> 5, d = o & 31;
#pragma unroll
      for (int r = 0; r < 8; ++r) {
        int nn = base_n + r + 8 * hi;
        k16[(((size_t)(b_ * 4 + head)) * N + nn) * 32 + d] = (_Float16)(acc[r] + bias);
      }
    } else {                                         // V stored transposed [b,h,d,N]
      int od = o - 128; int head = od >> 5, d = od & 31;
#pragma unroll
      for (int r = 0; r < 8; ++r) {
        int nn = base_n + r + 8 * hi;
        vT[(((size_t)(b_ * 4 + head)) * 32 + d) * N + nn] = (_Float16)(acc[r] + bias);
      }
    }
  }
}

// ---------------- k2: attention (two-pass softmax) ----------------
__global__ void __launch_bounds__(256)
attn_kernel(const _Float16* __restrict__ q16, const _Float16* __restrict__ k16,
            const _Float16* __restrict__ vT, const float* __restrict__ bias_table,
            const int* __restrict__ Hp, const int* __restrict__ Wp,
            float* __restrict__ attnOut, _Float16* __restrict__ oh16,
            int B_, int N) {
  __shared__ __align__(16) _Float16 plds[8 * 512];   // per-wave 16x32 f16 P tile

  int lane = threadIdx.x & 31, wid = threadIdx.x >> 5;
  int tpb = N >> 4;                                  // query tiles per (b,h)
  int gw = blockIdx.x * 8 + wid;
  if (gw >= B_ * 4 * tpb) return;                    // wave-uniform exit

  int bh = gw / tpb, qt = gw - bh * tpb;
  int h = bh & 3;
  int qbase = qt << 4;
  int H = *Hp, W = *Wp, W2 = 2 * W - 1;
  int nlo = lane & 15, hi = lane >> 4;

  const _Float16* qrows = q16 + (size_t)bh * N * 32;
  const _Float16* krows = k16 + (size_t)bh * N * 32;
  const _Float16* vrows = vT  + (size_t)bh * 32 * N;
  const float* btab = bias_table + h;                // stride 4 (NH = 4)
  v16h qfrag = load_frag(qrows, 32, qbase, 0, lane);

  // per-query-row bias constant: c_r = (qh+H-1)*W2 + qw+W-1
  int cr[8];
#pragma unroll
  for (int r = 0; r < 8; ++r) {
    int qr = qbase + r + 8 * hi;
    int qh = qr / W, qw = qr - qh * W;
    cr[r] = (qh + H - 1) * W2 + (qw + W - 1);
  }
  // per-lane key coord tracker: d_k = kh*W2 + kw, updated incrementally
  int kh0 = nlo / W, kw0 = nlo - kh0 * W;            // only key-coord divisions (none in K loops)
  const int dk_init = kh0 * W2 + kw0;
  const int bump = 16 + W - 1;                       // d_k delta on row wrap (W>=16)

  float m_run[8], l_run[8];
#pragma unroll
  for (int r = 0; r < 8; ++r) { m_run[r] = -1e30f; l_run[r] = 0.f; }

  const v8f Z = {0.f, 0.f, 0.f, 0.f, 0.f, 0.f, 0.f, 0.f};

  // ---- pass 1: lane-local online (m,l) over all keys, no shuffles ----
  {
    int kw = kw0, d_k = dk_init;
    for (int kb = 0; kb < N; kb += 16) {
      __builtin_prefetch(krows + (size_t)(kb + 16) * 32, 0, 3); // global_prefetch
      v16h kfrag = load_frag(krows, 32, kb, 0, lane);           // B = K tile
      v8f s = wmma16(qfrag, kfrag, Z);
#pragma unroll
      for (int r = 0; r < 8; ++r) {
        float v = s[r] + btab[(size_t)(cr[r] - d_k) * 4];
        float mnew = fmaxf(m_run[r], v);
        l_run[r] = l_run[r] * __expf(m_run[r] - mnew) + __expf(v - mnew);
        m_run[r] = mnew;
      }
      kw += 16;
      if (kw >= W) { kw -= W; d_k += bump; } else { d_k += 16; }
    }
  }
  // cross-lane merge of (m,l) across the 16 columns each half-wave owns
#pragma unroll
  for (int r = 0; r < 8; ++r) {
    float m = m_run[r], l = l_run[r];
#pragma unroll
    for (int mask = 1; mask <= 8; mask <<= 1) {
      float om = __shfl_xor(m, mask, 32);
      float ol = __shfl_xor(l, mask, 32);
      float mn = fmaxf(m, om);
      l = l * __expf(m - mn) + ol * __expf(om - mn);
      m = mn;
    }
    m_run[r] = m;
    l_run[r] = 1.0f / l;                             // store reciprocal
  }

  // ---- pass 2: write attn, accumulate O = P * V ----
  v8f o0 = Z, o1 = Z;
  float* attnRow = attnOut + ((size_t)bh * N + qbase) * N;
  const int lbase = wid * 512;

  int kw = kw0, d_k = dk_init;
  for (int kb = 0; kb < N; kb += 32) {
    v16h k0 = load_frag(krows, 32, kb,      0, lane);
    v16h k1 = load_frag(krows, 32, kb + 16, 0, lane);
    v8f s0 = wmma16(qfrag, k0, Z);
    v8f s1 = wmma16(qfrag, k1, Z);
#pragma unroll
    for (int t = 0; t < 2; ++t) {
      v8f s = t ? s1 : s0;
      int key = kb + t * 16 + nlo;
#pragma unroll
      for (int r = 0; r < 8; ++r) {
        float v = s[r] + btab[(size_t)(cr[r] - d_k) * 4];
        float p = __expf(v - m_run[r]) * l_run[r];
        attnRow[(size_t)(r + 8 * hi) * N + key] = p;            // softmax output
        plds[lbase + (r + 8 * hi) * 32 + t * 16 + nlo] = (_Float16)p;
      }
      kw += 16;
      if (kw >= W) { kw -= W; d_k += bump; } else { d_k += 16; }
    }
    // wave-private LDS: drain DS writes before cross-lane A-fragment reads
    __builtin_amdgcn_wave_barrier();
    asm volatile("s_wait_dscnt 0" ::: "memory");

    v16h pf;                                                    // P as A-fragment
    {
      int pb0 = lbase + nlo * 32 + hi * 8;
      v8h a = *(const v8h*)&plds[pb0];
      v8h b = *(const v8h*)&plds[pb0 + 16];
#pragma unroll
      for (int i = 0; i < 8; ++i) { pf[i] = a[i]; pf[i + 8] = b[i]; }
    }
    v16h vb0 = load_frag(vrows, N, 0,  kb, lane);               // V cols 0..15
    v16h vb1 = load_frag(vrows, N, 16, kb, lane);               // V cols 16..31
    o0 = wmma16(pf, vb0, o0);
    o1 = wmma16(pf, vb1, o1);
  }

  // store head output as f16 rows [b*N+n, h*32+d] for the projection GEMM
  int b = bh >> 2;
#pragma unroll
  for (int r = 0; r < 8; ++r) {
    int qr = qbase + r + 8 * hi;
    size_t ro = ((size_t)b * N + qr) * 128 + h * 32;
    oh16[ro + nlo]      = (_Float16)o0[r];
    oh16[ro + 16 + nlo] = (_Float16)o1[r];
  }
}

// ---------------- k3: output projection ----------------
__global__ void __launch_bounds__(256)
proj_kernel(const _Float16* __restrict__ oh16, const _Float16* __restrict__ pw16,
            const float* __restrict__ pb, float* __restrict__ outp, int tiles) {
  int lane = threadIdx.x & 31;
  int gw = (blockIdx.x * blockDim.x + threadIdx.x) >> 5;
  if (gw >= tiles) return;
  int tile_m = gw >> 3, tile_n = gw & 7;
  int row0 = tile_m * 16, col0 = tile_n * 16;

  v8f acc = {0.f, 0.f, 0.f, 0.f, 0.f, 0.f, 0.f, 0.f};
#pragma unroll
  for (int kc = 0; kc < 4; ++kc) {
    v16h a = load_frag(oh16, 128, row0, kc * 32, lane);
    v16h b = load_frag(pw16, 128, col0, kc * 32, lane);
    acc = wmma16(a, b, acc);
  }
  int hi = lane >> 4, nlo = lane & 15;
  int o = col0 + nlo;
  float bias = pb[o];
#pragma unroll
  for (int r = 0; r < 8; ++r)
    outp[(size_t)(row0 + r + 8 * hi) * 128 + o] = acc[r] + bias;
}

// ---------------- host ----------------
extern "C" void kernel_launch(void* const* d_in, const int* in_sizes, int n_in,
                              void* d_out, int out_size, void* d_ws, size_t ws_size,
                              hipStream_t stream) {
  const float* x          = (const float*)d_in[0];
  const float* q_w        = (const float*)d_in[1];
  const float* q_b        = (const float*)d_in[2];
  const float* kv_w       = (const float*)d_in[3];
  const float* kv_b       = (const float*)d_in[4];
  const float* proj_w     = (const float*)d_in[5];
  const float* proj_b     = (const float*)d_in[6];
  const float* bias_table = (const float*)d_in[7];
  // d_in[8] = rel_index: recomputed analytically on device, not read
  const int* Hp = (const int*)d_in[9];
  const int* Wp = (const int*)d_in[10];

  const int BN = in_sizes[0] / 128;                                // B*N
  const int N  = (int)std::lround(std::sqrt((double)in_sizes[8])); // rel_index is N*N
  const int B  = BN / N;
  const int n_qw = in_sizes[1], n_kvw = in_sizes[3], n_pw = in_sizes[5];

  // workspace carve-up (all 16B aligned)
  char* w = (char*)d_ws;
  _Float16* xh    = (_Float16*)w; w += (size_t)BN * 128 * 2;
  _Float16* qw16  = (_Float16*)w; w += (size_t)n_qw * 2;
  _Float16* kvw16 = (_Float16*)w; w += (size_t)n_kvw * 2;
  _Float16* pw16  = (_Float16*)w; w += (size_t)n_pw * 2;
  _Float16* q16   = (_Float16*)w; w += (size_t)BN * 128 * 2;
  _Float16* k16   = (_Float16*)w; w += (size_t)BN * 128 * 2;
  _Float16* vT16  = (_Float16*)w; w += (size_t)BN * 128 * 2;
  _Float16* oh16  = (_Float16*)w; w += (size_t)BN * 128 * 2;

  float* outp  = (float*)d_out;
  float* attnp = outp + (size_t)BN * 128;

  // k0: conversions
  {
    int total = BN * 128 + n_qw + n_kvw + n_pw;
    convert_kernel<<<(total + 255) / 256, 256, 0, stream>>>(
        x, q_w, kv_w, proj_w, xh, qw16, kvw16, pw16, BN * 128, n_qw, n_kvw, n_pw);
  }
  // k1: QKV projection
  {
    int tiles_q = (BN / 16) * 8;
    int tiles_total = tiles_q + (BN / 16) * 16;
    qkv_kernel<<<(tiles_total + 7) / 8, 256, 0, stream>>>(
        xh, qw16, kvw16, q_b, kv_b, q16, k16, vT16, BN, N, tiles_q, tiles_total);
  }
  // k2: attention
  {
    int waves = B * 4 * (N / 16);
    attn_kernel<<<(waves + 7) / 8, 256, 0, stream>>>(
        q16, k16, vT16, bias_table, Hp, Wp, attnp, oh16, B, N);
  }
  // k3: output projection
  {
    int tiles = (BN / 16) * 8;
    proj_kernel<<<(tiles + 7) / 8, 256, 0, stream>>>(oh16, pw16, proj_b, outp, tiles);
  }
}
